// PrimedGDNLayer_16690242912582
// MI455X (gfx1250) — compile-verified, matrix operand order checked by reading, not run on hardware
//
#include <hip/hip_runtime.h>
#include <hip/hip_bf16.h>

// ---------------- problem constants (from reference) ----------------
static constexpr int Bc    = 2;
static constexpr int Tc    = 1024;
static constexpr int HID   = 1024;
static constexpr int NQ    = 8;
static constexpr int NK    = 4;
static constexpr int DK    = 128;
static constexpr int DV    = 128;
static constexpr int KDIM  = NQ * DK;            // 1024
static constexpr int KKDIM = NK * DK;            // 512
static constexpr int KVDIM = NK * DV;            // 512
static constexpr int VDIM  = NQ * DV;            // 1024
static constexpr int CONVD = KDIM + KKDIM + KVDIM;           // 2048
static constexpr int TOT   = CONVD + VDIM + NQ + NQ;         // 3088
static constexpr int KSZ   = 4;
static constexpr int BT    = Bc * Tc;            // 2048

typedef __attribute__((ext_vector_type(16))) __bf16 v16bf;
typedef __attribute__((ext_vector_type(8)))  __bf16 v8bf;
typedef __attribute__((ext_vector_type(8)))  float  v8f;

// ---------------- fp32 -> bf16 conversion ----------------
__global__ void k_cvt_bf16(const float* __restrict__ in, __bf16* __restrict__ out, int n) {
    int i = blockIdx.x * blockDim.x + threadIdx.x;
    if (i < n) out[i] = (__bf16)in[i];
}

// in: [R, C] fp32 -> out: [C, R] bf16   (transpose so GEMM B-fragments are K-contiguous)
__global__ void k_cvt_bf16_T(const float* __restrict__ in, __bf16* __restrict__ out,
                             int R, int C) {
    int i = blockIdx.x * blockDim.x + threadIdx.x;
    if (i < R * C) {
        int r = i / C, c = i % C;
        out[(size_t)c * R + r] = (__bf16)in[(size_t)r * C + c];
    }
}

// ---------------- WMMA bf16 GEMM: C[M,N] = A[M,K] * B[K,N], BTm = B^T [N,K] ----------------
// One wave computes a 64x16 tile (4 stacked 16x16 accumulators): the B (weight)
// fragment is loaded once per 32-deep K step and reused across 4
// v_wmma_f32_16x16x32_bf16 issues -> ~12.8 FLOP/byte, 4x weight-traffic reuse.
// Tile predicate is wave-uniform so EXEC stays all-ones around WMMA.
// Requires M % 64 == 0 and N % 16 == 0 (true for both GEMMs here).
__global__ void k_wmma_gemm_bf16(const __bf16* __restrict__ A,
                                 const __bf16* __restrict__ BTm,
                                 float* __restrict__ C,
                                 int M, int N, int K) {
    const int lane   = threadIdx.x & 31;
    const int wave   = threadIdx.x >> 5;
    const int tilesN = N >> 4;
    const int tile   = blockIdx.x * (blockDim.x >> 5) + wave;
    const int nTiles = (M >> 6) * tilesN;
    if (tile >= nTiles) return;                    // wave-uniform

    const int tm = (tile / tilesN) << 6;           // 64-row block
    const int tn = (tile % tilesN) << 4;           // 16-col block
    const int r  = lane & 15;
    const int hi = (lane >> 4) & 1;                // half-wave selects K phase

    const __bf16* Arow0 = A + (size_t)(tm +  0 + r) * K;
    const __bf16* Arow1 = A + (size_t)(tm + 16 + r) * K;
    const __bf16* Arow2 = A + (size_t)(tm + 32 + r) * K;
    const __bf16* Arow3 = A + (size_t)(tm + 48 + r) * K;
    const __bf16* Brow  = BTm + (size_t)(tn + r) * K;

    v8f acc0 = {}, acc1 = {}, acc2 = {}, acc3 = {};
    for (int kb = 0; kb < K; kb += 32) {
        const int k0 = kb + hi * 8;                // lanes 0-15: K 0-7/16-23 ; 16-31: 8-15/24-31
        union { v16bf v; v8bf h[2]; } b, a0, a1, a2, a3;
        b.h[0]  = *(const v8bf*)(Brow  + k0);
        b.h[1]  = *(const v8bf*)(Brow  + k0 + 16);
        a0.h[0] = *(const v8bf*)(Arow0 + k0);
        a0.h[1] = *(const v8bf*)(Arow0 + k0 + 16);
        a1.h[0] = *(const v8bf*)(Arow1 + k0);
        a1.h[1] = *(const v8bf*)(Arow1 + k0 + 16);
        a2.h[0] = *(const v8bf*)(Arow2 + k0);
        a2.h[1] = *(const v8bf*)(Arow2 + k0 + 16);
        a3.h[0] = *(const v8bf*)(Arow3 + k0);
        a3.h[1] = *(const v8bf*)(Arow3 + k0 + 16);
        acc0 = __builtin_amdgcn_wmma_f32_16x16x32_bf16(false, a0.v, false, b.v, (short)0, acc0, false, false);
        acc1 = __builtin_amdgcn_wmma_f32_16x16x32_bf16(false, a1.v, false, b.v, (short)0, acc1, false, false);
        acc2 = __builtin_amdgcn_wmma_f32_16x16x32_bf16(false, a2.v, false, b.v, (short)0, acc2, false, false);
        acc3 = __builtin_amdgcn_wmma_f32_16x16x32_bf16(false, a3.v, false, b.v, (short)0, acc3, false, false);
    }
    // C/D layout per accumulator: VGPR i = row (sub_tm + i + 8*hi), column tn + (lane&15)
    {
        float* Cp = C + (size_t)(tm +  0 + hi * 8) * N + (tn + r);
        #pragma unroll
        for (int i = 0; i < 8; ++i) Cp[(size_t)i * N] = acc0[i];
    }
    {
        float* Cp = C + (size_t)(tm + 16 + hi * 8) * N + (tn + r);
        #pragma unroll
        for (int i = 0; i < 8; ++i) Cp[(size_t)i * N] = acc1[i];
    }
    {
        float* Cp = C + (size_t)(tm + 32 + hi * 8) * N + (tn + r);
        #pragma unroll
        for (int i = 0; i < 8; ++i) Cp[(size_t)i * N] = acc2[i];
    }
    {
        float* Cp = C + (size_t)(tm + 48 + hi * 8) * N + (tn + r);
        #pragma unroll
        for (int i = 0; i < 8; ++i) Cp[(size_t)i * N] = acc3[i];
    }
}

// ---------------- causal depthwise conv (KSZ=4) + SiLU over first CONVD channels ----------------
__global__ void k_conv_silu(const float* __restrict__ mixed,
                            const float* __restrict__ conv_w,
                            float* __restrict__ qkv) {
    int i = blockIdx.x * blockDim.x + threadIdx.x;       // over BT*CONVD
    if (i >= BT * CONVD) return;
    int c  = i % CONVD;
    int bt = i / CONVD;
    int t  = bt % Tc;
    float s = 0.f;
    #pragma unroll
    for (int j = 0; j < KSZ; ++j) {
        int tt = t + j - (KSZ - 1);
        if (tt >= 0)
            s += mixed[(size_t)(bt + (tt - t)) * TOT + c] * conv_w[c * KSZ + j];
    }
    qkv[(size_t)bt * CONVD + c] = s / (1.f + __expf(-s));  // silu
}

// ---------------- q/k L2-norm + gate prep: one wave per (bt, head-slot) ----------------
// slots 0..7 = q heads, 8..11 = k heads
__global__ void k_prep(const float* __restrict__ qkv, const float* __restrict__ mixed,
                       const float* __restrict__ A_log, const float* __restrict__ dt_bias,
                       float* __restrict__ qn, float* __restrict__ kn,
                       float* __restrict__ gb, float* __restrict__ betab) {
    int wid  = (blockIdx.x * blockDim.x + threadIdx.x) >> 5;
    int lane = threadIdx.x & 31;
    if (wid >= BT * 12) return;
    int h  = wid % 12;
    int bt = wid / 12;

    const float* src;
    float* dst;
    if (h < 8) { src = qkv + (size_t)bt * CONVD + h * DK;
                 dst = qn  + (size_t)bt * KDIM  + h * DK; }
    else       { src = qkv + (size_t)bt * CONVD + KDIM + (h - 8) * DK;
                 dst = kn  + (size_t)bt * KKDIM + (h - 8) * DK; }

    float v[4]; float ss = 0.f;
    #pragma unroll
    for (int j = 0; j < 4; ++j) { v[j] = src[lane + j * 32]; ss += v[j] * v[j]; }
    #pragma unroll
    for (int off = 16; off > 0; off >>= 1) ss += __shfl_xor(ss, off, 32);
    float rn = rsqrtf(ss + 1e-6f);
    if (h < 8) rn *= 0.08838834764831845f;               // * DK^-0.5
    #pragma unroll
    for (int j = 0; j < 4; ++j) dst[lane + j * 32] = v[j] * rn;

    if (h < 8 && lane == 0) {
        float aval = mixed[(size_t)bt * TOT + (CONVD + VDIM + NQ) + h];  // a
        float bval = mixed[(size_t)bt * TOT + (CONVD + VDIM) + h];       // b
        float sp   = log1pf(__expf(aval + dt_bias[h]));                  // softplus
        gb[bt * NQ + h]    = -__expf(A_log[h]) * sp;
        betab[bt * NQ + h] = 1.f / (1.f + __expf(-bval));
    }
}

// ---------------- gated delta-rule scan: 1 block per (b,h), state column in VGPRs ----------------
__global__ __launch_bounds__(128) void k_scan(const float* __restrict__ qn,
                                              const float* __restrict__ kn,
                                              const float* __restrict__ qkv,
                                              const float* __restrict__ gb,
                                              const float* __restrict__ betab,
                                              float* __restrict__ o) {
    __shared__ float lds_k[DK];
    __shared__ float lds_q[DK];
    const int bh  = blockIdx.x;           // 0..15
    const int b   = bh >> 3, h = bh & 7, kvh = h >> 1;   // GQA: repeat factor 2
    const int v   = threadIdx.x;          // column index 0..127

    float S[DK];
    #pragma unroll
    for (int k = 0; k < DK; ++k) S[k] = 0.f;

    for (int t = 0; t < Tc; ++t) {
        const size_t bt = (size_t)b * Tc + t;
        lds_k[v] = kn[bt * KKDIM + kvh * DK + v];
        lds_q[v] = qn[bt * KDIM  + h   * DK + v];
        float vt   = qkv[bt * CONVD + KDIM + KKDIM + kvh * DV + v];
        float eg   = __expf(gb[bt * NQ + h]);
        float beta = betab[bt * NQ + h];
        __syncthreads();

        float kv = 0.f;
        #pragma unroll
        for (int k = 0; k < DK; ++k) kv += lds_k[k] * S[k];
        kv *= eg;                                  // dot with decayed state
        float u = beta * (vt - kv);

        float ov = 0.f;
        #pragma unroll
        for (int k = 0; k < DK; ++k) {
            S[k] = S[k] * eg + lds_k[k] * u;       // decay + rank-1 update
            ov  += lds_q[k] * S[k];                // output with new state
        }
        o[(bt * NQ + h) * DV + v] = ov;
        __syncthreads();
    }
}

// ---------------- gated RMSNorm * silu(z), emit bf16 for final GEMM: 1 wave per (bt,h) ----------------
__global__ void k_norm_gate(const float* __restrict__ o, const float* __restrict__ mixed,
                            const float* __restrict__ norm_w, __bf16* __restrict__ ybf) {
    int wid  = (blockIdx.x * blockDim.x + threadIdx.x) >> 5;
    int lane = threadIdx.x & 31;
    if (wid >= BT * NQ) return;
    int h = wid % NQ;
    size_t bt = wid / NQ;
    const float* op = o + (bt * NQ + h) * DV;

    float vals[4]; float ss = 0.f;
    #pragma unroll
    for (int j = 0; j < 4; ++j) { vals[j] = op[lane + j * 32]; ss += vals[j] * vals[j]; }
    #pragma unroll
    for (int off = 16; off > 0; off >>= 1) ss += __shfl_xor(ss, off, 32);
    float r = rsqrtf(ss * (1.f / DV) + 1e-5f);

    #pragma unroll
    for (int j = 0; j < 4; ++j) {
        int idx = lane + j * 32;
        float z = mixed[bt * TOT + CONVD + h * DV + idx];
        float y = vals[j] * r * norm_w[idx] * (z / (1.f + __expf(-z)));
        ybf[bt * VDIM + h * DV + idx] = (__bf16)y;
    }
}

// ---------------- host-side orchestration ----------------
extern "C" void kernel_launch(void* const* d_in, const int* in_sizes, int n_in,
                              void* d_out, int out_size, void* d_ws, size_t ws_size,
                              hipStream_t stream) {
    const float* x       = (const float*)d_in[0];  // [B,T,HID]
    const float* w_in    = (const float*)d_in[1];  // [HID,TOT]
    const float* conv_w  = (const float*)d_in[2];  // [CONVD,KSZ]
    const float* A_log   = (const float*)d_in[3];  // [NQ]
    const float* dt_bias = (const float*)d_in[4];  // [NQ]
    const float* norm_w  = (const float*)d_in[5];  // [DV]
    const float* w_out   = (const float*)d_in[6];  // [VDIM,HID]
    float* out = (float*)d_out;                    // [B,T,HID]

    // workspace carve-up (256B aligned)
    char* ws = (char*)d_ws;
    size_t off = 0;
    auto carve = [&](size_t bytes) { void* p = ws + off; off = (off + bytes + 255) & ~(size_t)255; return p; };
    __bf16* xbf    = (__bf16*)carve((size_t)BT * HID * 2);
    __bf16* winT   = (__bf16*)carve((size_t)TOT * HID * 2);
    __bf16* woutT  = (__bf16*)carve((size_t)HID * VDIM * 2);
    float*  mixed  = (float*) carve((size_t)BT * TOT * 4);
    float*  qkv    = (float*) carve((size_t)BT * CONVD * 4);
    float*  qn     = (float*) carve((size_t)BT * KDIM * 4);
    float*  kn     = (float*) carve((size_t)BT * KKDIM * 4);
    float*  gbuf   = (float*) carve((size_t)BT * NQ * 4);
    float*  bbuf   = (float*) carve((size_t)BT * NQ * 4);
    float*  obuf   = (float*) carve((size_t)BT * VDIM * 4);
    __bf16* ybf    = (__bf16*)carve((size_t)BT * VDIM * 2);

    // 1) bf16 conversions (x, w_in^T, w_out^T)
    {
        int n = BT * HID;
        k_cvt_bf16<<<(n + 255) / 256, 256, 0, stream>>>(x, xbf, n);
    }
    {
        int n = HID * TOT;
        k_cvt_bf16_T<<<(n + 255) / 256, 256, 0, stream>>>(w_in, winT, HID, TOT);
    }
    {
        int n = VDIM * HID;
        k_cvt_bf16_T<<<(n + 255) / 256, 256, 0, stream>>>(w_out, woutT, VDIM, HID);
    }

    // 2) mixed = x @ w_in   (WMMA bf16, f32 accum)  M=2048 N=3088 K=1024
    {
        int tiles = (BT / 64) * (TOT / 16);        // 64x16 per wave
        k_wmma_gemm_bf16<<<(tiles + 7) / 8, 256, 0, stream>>>(xbf, winT, mixed, BT, TOT, HID);
    }

    // 3) depthwise causal conv + silu
    {
        int n = BT * CONVD;
        k_conv_silu<<<(n + 255) / 256, 256, 0, stream>>>(mixed, conv_w, qkv);
    }

    // 4) l2-norm q/k, gates
    {
        int waves = BT * 12;
        k_prep<<<(waves * 32 + 255) / 256, 256, 0, stream>>>(qkv, mixed, A_log, dt_bias,
                                                             qn, kn, gbuf, bbuf);
    }

    // 5) sequential gated delta-rule scan (16 blocks, register-resident state)
    k_scan<<<Bc * NQ, 128, 0, stream>>>(qn, kn, qkv, gbuf, bbuf, obuf);

    // 6) gated RMSNorm -> bf16
    {
        int waves = BT * NQ;
        k_norm_gate<<<(waves * 32 + 255) / 256, 256, 0, stream>>>(obuf, mixed, norm_w, ybf);
    }

    // 7) out = y @ w_out   (WMMA bf16)  M=2048 N=1024 K=1024
    {
        int tiles = (BT / 64) * (HID / 16);        // 64x16 per wave
        k_wmma_gemm_bf16<<<(tiles + 7) / 8, 256, 0, stream>>>(ybf, woutT, out, BT, HID, VDIM);
    }
}
